// HGT_70712341561905
// MI455X (gfx1250) — matrix operand order repaired and checked by reading, not compiled, not run on hardware
//
#include <hip/hip_runtime.h>

// ---------------- constants ----------------
constexpr int cNI = 60000, cNP = 30000;
constexpr int cEII = 480000, cEIP = 120000, cEPI = 120000;
constexpr int cHID = 128, cHEADS = 4;
constexpr int cKPOOL = 8;
constexpr float cSCALE = 0.17677669529663687f; // 1/sqrt(32)

typedef __attribute__((ext_vector_type(16))) _Float16 v16h;
typedef __attribute__((ext_vector_type(8)))  float    v8f;

// ---------------- device helpers ----------------
__device__ __forceinline__ float gelu_f(float x) {
    return 0.5f * x * (1.0f + erff(x * 0.7071067811865475f));
}
__device__ __forceinline__ unsigned fenc(float x) {
    unsigned u = __float_as_uint(x);
    return (u & 0x80000000u) ? ~u : (u | 0x80000000u);
}
__device__ __forceinline__ float fdec(unsigned u) {
    unsigned v = (u & 0x80000000u) ? (u & 0x7FFFFFFFu) : ~u;
    return __uint_as_float(v);
}
__device__ __forceinline__ float wave_sum32(float s) {
    #pragma unroll
    for (int m = 16; m >= 1; m >>= 1) s += __shfl_xor(s, m, 32);
    return s;
}

// ---------------- generic utility kernels ----------------
__global__ void k_fill(float* __restrict__ p, float v, size_t n) {
    size_t i = (size_t)blockIdx.x * blockDim.x + threadIdx.x;
    if (i < n) p[i] = v;
}
__global__ void k_copy(float* __restrict__ d, const float* __restrict__ s, size_t n) {
    size_t i = (size_t)blockIdx.x * blockDim.x + threadIdx.x;
    if (i < n) d[i] = s[i];
}
__global__ void k_gelu(const float* __restrict__ in, float* __restrict__ out, size_t n) {
    size_t i = (size_t)blockIdx.x * blockDim.x + threadIdx.x;
    if (i < n) out[i] = gelu_f(in[i]);
}
// t = beta*t + (1-beta)*x   (gated skip; beta = sigmoid(skip))
__global__ void k_skip(float* __restrict__ t, const float* __restrict__ x,
                       const float* __restrict__ skipP, size_t n) {
    size_t i = (size_t)blockIdx.x * blockDim.x + threadIdx.x;
    if (i >= n) return;
    float beta = 1.0f / (1.0f + __expf(-skipP[0]));
    t[i] = beta * t[i] + (1.0f - beta) * x[i];
}

// ---------------- WMMA GEMM:  Y = act(X @ W + b) ----------------
// f32 in/out, f16 compute, f32 accumulate.  One wave = one 16x16 output tile.
// gridDim.y batches heads via element offsets (xHead/wHead/yHead).
__global__ void __launch_bounds__(32)
k_wmma_gemm(const float* __restrict__ X, int lda,
            const float* __restrict__ Wm, int ldw,
            const float* __restrict__ bias,
            float* __restrict__ Y, int ldy,
            int M, int K, int Nout, int act,
            int xHead, int wHead, int yHead)
{
    X  += (size_t)blockIdx.y * xHead;
    Wm += (size_t)blockIdx.y * wHead;
    Y  += (size_t)blockIdx.y * yHead;
    const int ntn = Nout >> 4;
    const int tm = blockIdx.x / ntn;
    const int tn = blockIdx.x % ntn;
    const int lane = threadIdx.x;
    const int hf  = lane >> 4;   // 0/1 half-wave
    const int l16 = lane & 15;
    int rowA = tm * 16 + l16; if (rowA >= M) rowA = M - 1;  // clamp: keep EXEC full
    const int colB = tn * 16 + l16;
    v8f acc = {};
    for (int k0 = 0; k0 < K; k0 += 32) {
        // A tile (16x32 f16): lane<16 rows M=l16 K=[k0+0..7],[k0+16..23]; lane>=16 K+8
        const float* xr = X + (size_t)rowA * lda + k0 + hf * 8;
        v16h a;
        #pragma unroll
        for (int j = 0; j < 8; ++j) a[j]     = (_Float16)xr[j];
        #pragma unroll
        for (int j = 0; j < 8; ++j) a[8 + j] = (_Float16)xr[16 + j];
        // B tile (32x16 f16): lane<16 K=[k0+0..7],[k0+16..23]; lane>=16 K+8, N=l16
        const float* wr = Wm + (size_t)(k0 + hf * 8) * ldw + colB;
        v16h b;
        #pragma unroll
        for (int e = 0; e < 16; ++e) b[e] = (_Float16)wr[(size_t)(e + (e & 8)) * ldw];
        acc = __builtin_amdgcn_wmma_f32_16x16x32_f16(false, a, false, b,
                                                     (short)0, acc, false, false);
    }
    const float bv = bias ? bias[colB] : 0.0f;
    #pragma unroll
    for (int r = 0; r < 8; ++r) {
        int row = tm * 16 + hf * 8 + r;   // C/D layout: VGPR r -> M = r + 8*half
        if (row < M) {
            float v = acc[r] + bv;
            if (act) v = gelu_f(v);
            Y[(size_t)row * ldy + colB] = v;
        }
    }
}

// ---------------- HGT edge passes (segment softmax over dst, per relation) ----
// Pass A: logits + atomic-max per (dst,head).  One wave per edge.
__global__ void k_edge_logits(const int* __restrict__ src, const int* __restrict__ dst, int E,
                              const float* __restrict__ ke, const float* __restrict__ q,
                              const float* __restrict__ prel, float scale,
                              float* __restrict__ al, unsigned* __restrict__ mstat)
{
    int e = (int)(((size_t)blockIdx.x * blockDim.x + threadIdx.x) >> 5);
    if (e >= E) return;
    int lane = threadIdx.x & 31;
    int h = lane >> 3, t = lane & 7;           // 8 lanes per head
    int s = src[e], d = dst[e];
    const float* qp = q  + (size_t)d * cHID + h * 32;
    const float* kp = ke + (size_t)s * cHID + h * 32;
    float p = 0.0f;
    #pragma unroll
    for (int j = t; j < 32; j += 8) p += qp[j] * kp[j];
    p += __shfl_xor(p, 1, 32); p += __shfl_xor(p, 2, 32); p += __shfl_xor(p, 4, 32);
    float lg = p * prel[h] * scale;
    if (t == 0) {
        al[(size_t)e * 4 + h] = lg;
        atomicMax(&mstat[(size_t)d * 4 + h], fenc(lg));
    }
}
// Pass B: e = exp(l - max); atomic sum per (dst,head).  One thread per (edge,head).
__global__ void k_edge_exp(const int* __restrict__ dst, int E,
                           float* __restrict__ al, const unsigned* __restrict__ mstat,
                           float* __restrict__ sstat)
{
    size_t i = (size_t)blockIdx.x * blockDim.x + threadIdx.x;
    if (i >= (size_t)E * 4) return;
    int e = (int)(i >> 2), h = (int)(i & 3);
    int d = dst[e];
    float m  = fdec(mstat[(size_t)d * 4 + h]);
    float ex = __expf(al[i] - m);
    al[i] = ex;
    atomicAdd(&sstat[(size_t)d * 4 + h], ex);
}
// Pass C: out[dst] += (e/sum) * ve[src].  One wave per edge, 128 atomics.
__global__ void k_edge_scatter(const int* __restrict__ src, const int* __restrict__ dst, int E,
                               const float* __restrict__ al, const float* __restrict__ sstat,
                               const float* __restrict__ ve, float* __restrict__ out)
{
    int e = (int)(((size_t)blockIdx.x * blockDim.x + threadIdx.x) >> 5);
    if (e >= E) return;
    int lane = threadIdx.x & 31;
    int h = lane >> 3, t = lane & 7;
    int s = src[e], d = dst[e];
    float coeff = al[(size_t)e * 4 + h] / (sstat[(size_t)d * 4 + h] + 1e-16f);
    const float* v = ve + (size_t)s * cHID + h * 32;
    float* o = out + (size_t)d * cHID + h * 32;
    #pragma unroll
    for (int j = t; j < 32; j += 8) atomicAdd(&o[j], coeff * v[j]);
}

// ---------------- LayerNorm (+opt residual) + gelu; one wave per row of 128 ---
__global__ void k_ln_gelu(const float* __restrict__ in, const float* __restrict__ w,
                          const float* __restrict__ b, const float* __restrict__ res,
                          float* __restrict__ out, int M)
{
    int row = (int)(((size_t)blockIdx.x * blockDim.x + threadIdx.x) >> 5);
    if (row >= M) return;
    int lane = threadIdx.x & 31;
    const float* x = in + (size_t)row * cHID;
    float v[4]; float s = 0.0f;
    #pragma unroll
    for (int j = 0; j < 4; ++j) { v[j] = x[lane + 32 * j]; s += v[j]; }
    s = wave_sum32(s);
    float mu = s * (1.0f / 128.0f);
    float q = 0.0f;
    #pragma unroll
    for (int j = 0; j < 4; ++j) { float d = v[j] - mu; q += d * d; }
    q = wave_sum32(q);
    float inv = rsqrtf(q * (1.0f / 128.0f) + 1e-5f);
    float* o = out + (size_t)row * cHID;
    const float* r = res ? res + (size_t)row * cHID : nullptr;
    #pragma unroll
    for (int j = 0; j < 4; ++j) {
        int c = lane + 32 * j;
        float y = (v[j] - mu) * inv * w[c] + b[c];
        if (r) y += r[c];
        o[c] = gelu_f(y);
    }
}

// ---------------- GAT pooling stage --------------------------------------------
__global__ void k_node_dot(const float* __restrict__ xh, const float* __restrict__ w,
                           float* __restrict__ xg, int N)
{
    int n = (int)(((size_t)blockIdx.x * blockDim.x + threadIdx.x) >> 5);
    if (n >= N) return;
    int lane = threadIdx.x & 31;
    const float* x = xh + (size_t)n * cHID;
    float s = 0.0f;
    #pragma unroll
    for (int j = 0; j < 4; ++j) s += x[lane + 32 * j] * w[lane + 32 * j];
    s = wave_sum32(s);
    if (lane == 0) xg[n] = s;
}
__global__ void k_gat_logit(const int* src, const int* dst, int soff, int doff, int E, int isLoop,
                            const float* __restrict__ xg, const float* aS, const float* aD,
                            float* __restrict__ al, unsigned* __restrict__ mg)
{
    int e = (int)((size_t)blockIdx.x * blockDim.x + threadIdx.x);
    if (e >= E) return;
    int s, d;
    if (isLoop) { s = e; d = e; } else { s = src[e] + soff; d = dst[e] + doff; }
    float v = xg[s] * aS[0] + xg[d] * aD[0];
    v = (v > 0.0f) ? v : 0.2f * v;            // leaky_relu(0.2)
    al[e] = v;
    atomicMax(&mg[d], fenc(v));
}
__global__ void k_gat_exp(const int* src, const int* dst, int soff, int doff, int E, int isLoop,
                          float* __restrict__ al, const unsigned* __restrict__ mg,
                          float* __restrict__ sg)
{
    int e = (int)((size_t)blockIdx.x * blockDim.x + threadIdx.x);
    if (e >= E) return;
    int d = isLoop ? e : (dst[e] + doff);
    (void)src; (void)soff;
    float m = fdec(mg[d]);
    float ex = __expf(al[e] - m);
    al[e] = ex;
    atomicAdd(&sg[d], ex);
}
__global__ void k_gat_score(const int* src, const int* dst, int soff, int doff, int E, int isLoop,
                            const float* __restrict__ xg, const float* __restrict__ al,
                            const float* __restrict__ sg, float* __restrict__ sc)
{
    int e = (int)((size_t)blockIdx.x * blockDim.x + threadIdx.x);
    if (e >= E) return;
    int s, d;
    if (isLoop) { s = e; d = e; } else { s = src[e] + soff; d = dst[e] + doff; }
    float coeff = al[e] / (sg[d] + 1e-16f);
    atomicAdd(&sc[d], coeff * xg[s]);
}
// top-8 with jax tie semantics (value desc, then smallest index); single block.
__global__ void k_topk8(const float* __restrict__ score, int N, const float* biasP,
                        int* __restrict__ idx_out, float* __restrict__ tv_out)
{
    __shared__ float sv[256 * 8];
    __shared__ int   si[256 * 8];
    int tid = threadIdx.x;
    float bias = biasP[0];
    float lv[8]; int li[8];
    #pragma unroll
    for (int j = 0; j < 8; ++j) { lv[j] = -__builtin_inff(); li[j] = 0x7fffffff; }
    for (int n = tid; n < N; n += 256) {
        float v = score[n] + bias;
        if (v > lv[7]) {
            int j = 7;
            while (j > 0 && v > lv[j - 1]) { lv[j] = lv[j - 1]; li[j] = li[j - 1]; --j; }
            lv[j] = v; li[j] = n;
        }
    }
    #pragma unroll
    for (int j = 0; j < 8; ++j) { sv[tid * 8 + j] = lv[j]; si[tid * 8 + j] = li[j]; }
    __syncthreads();
    if (tid == 0) {
        for (int k = 0; k < 8; ++k) {
            float best = -__builtin_inff(); int bi = 0x7fffffff; int bp = 0;
            for (int c = 0; c < 2048; ++c) {
                float v = sv[c];
                if (v > best || (v == best && si[c] < bi)) { best = v; bi = si[c]; bp = c; }
            }
            sv[bp] = -__builtin_inff();
            tv_out[k] = tanhf(best);
            idx_out[k] = bi;
        }
    }
}
__global__ void k_pool_gather(const float* __restrict__ xh, const int* __restrict__ idx,
                              const float* __restrict__ tv, float* __restrict__ hb)
{
    int t = blockIdx.x * blockDim.x + threadIdx.x;  // 1024 threads
    if (t >= cKPOOL * cHID) return;
    int k = t >> 7, c = t & 127;
    hb[t] = xh[(size_t)idx[k] * cHID + c] * tv[k];
}
// fc1(1024->128) gelu, fc2(128->64) gelu, fc3(64->1) gelu, nan_to_num. 128 thr.
__global__ void k_head_mlp(const float* __restrict__ h,
                           const float* w1, const float* b1,
                           const float* w2, const float* b2,
                           const float* w3, const float* b3,
                           float* __restrict__ out)
{
    __shared__ float s1[128];
    __shared__ float s2[64];
    int tid = threadIdx.x;
    float acc = b1[tid];
    for (int i = 0; i < 1024; ++i) acc += h[i] * w1[i * 128 + tid];
    s1[tid] = gelu_f(acc);
    __syncthreads();
    if (tid < 64) {
        float a = b2[tid];
        for (int i = 0; i < 128; ++i) a += s1[i] * w2[i * 64 + tid];
        s2[tid] = gelu_f(a);
    }
    __syncthreads();
    if (tid == 0) {
        float a = b3[0];
        for (int i = 0; i < 64; ++i) a += s2[i] * w3[i];
        a = gelu_f(a);
        if (isnan(a)) a = 0.0f;
        else if (isinf(a)) a = (a > 0.0f) ? 3.4028235e38f : -3.4028235e38f;
        out[0] = a;
    }
}

// ---------------- host side ----------------
struct ConvP {
    const float *a_b[2], *a_rel[3], *a_w[2], *k_b[2], *k_w[2], *m_rel[3],
                *p_rel[3], *q_b[2], *q_w[2], *skip[2], *v_b[2], *v_w[2];
};

extern "C" void kernel_launch(void* const* d_in, const int* in_sizes, int n_in,
                              void* d_out, int out_size, void* d_ws, size_t ws_size,
                              hipStream_t stream)
{
    (void)in_sizes; (void)n_in; (void)out_size;
    const float* x_instr = (const float*)d_in[0];
    const float* x_port  = (const float*)d_in[1];
    const int* ei_ii = (const int*)d_in[2];
    const int* ei_ip = (const int*)d_in[3];
    const int* ei_pi = (const int*)d_in[4];
    // params flattened: top-level insertion order, nested dicts alphabetical
    int c = 5;
    ConvP conv[3];
    for (int l = 0; l < 3; ++l) {
        ConvP& P = conv[l];
        for (int t = 0; t < 2; ++t) P.a_b[t]   = (const float*)d_in[c++];
        for (int r = 0; r < 3; ++r) P.a_rel[r] = (const float*)d_in[c++];
        for (int t = 0; t < 2; ++t) P.a_w[t]   = (const float*)d_in[c++];
        for (int t = 0; t < 2; ++t) P.k_b[t]   = (const float*)d_in[c++];
        for (int t = 0; t < 2; ++t) P.k_w[t]   = (const float*)d_in[c++];
        for (int r = 0; r < 3; ++r) P.m_rel[r] = (const float*)d_in[c++];
        for (int r = 0; r < 3; ++r) P.p_rel[r] = (const float*)d_in[c++];
        for (int t = 0; t < 2; ++t) P.q_b[t]   = (const float*)d_in[c++];
        for (int t = 0; t < 2; ++t) P.q_w[t]   = (const float*)d_in[c++];
        for (int t = 0; t < 2; ++t) P.skip[t]  = (const float*)d_in[c++];
        for (int t = 0; t < 2; ++t) P.v_b[t]   = (const float*)d_in[c++];
        for (int t = 0; t < 2; ++t) P.v_w[t]   = (const float*)d_in[c++];
    }
    const float* fc1_b = (const float*)d_in[c++]; const float* fc1_w = (const float*)d_in[c++];
    const float* fc2_b = (const float*)d_in[c++]; const float* fc2_w = (const float*)d_in[c++];
    const float* fc3_b = (const float*)d_in[c++]; const float* fc3_w = (const float*)d_in[c++];
    const float* g_adst = (const float*)d_in[c++]; const float* g_asrc = (const float*)d_in[c++];
    const float* g_bias = (const float*)d_in[c++]; const float* g_w    = (const float*)d_in[c++];
    const float* nrm[2][2][2]; // [type][layer][0=b,1=w]
    for (int t = 0; t < 2; ++t)
        for (int l = 0; l < 2; ++l) { nrm[t][l][0] = (const float*)d_in[c++];
                                      nrm[t][l][1] = (const float*)d_in[c++]; }

    // workspace carve
    const size_t SI = (size_t)cNI * cHID, SP = (size_t)cNP * cHID;
    float* W = (float*)d_ws;
    auto carve = [&](size_t n) { float* p = W; W += n; return p; };
    float* xi   = carve(SI);  float* xp_  = carve(SP);
    float* ki   = carve(SI);  float* qi   = carve(SI);  float* vi = carve(SI);
    float* kp   = carve(SP);  float* qp   = carve(SP);  float* vp = carve(SP);
    float* keii = carve(SI);  float* veii = carve(SI);
    float* keip = carve(SI);  float* veip = carve(SI);
    float* kepi = carve(SP);  float* vepi = carve(SP);
    float* oi   = carve(SI);  float* op_  = carve(SP);
    float* ti   = carve(SI);  float* tp   = carve(SP);
    float* al   = carve((size_t)cEII * cHEADS);
    float* msI  = carve((size_t)cNI * cHEADS);  float* ssI = carve((size_t)cNI * cHEADS);
    float* msP  = carve((size_t)cNP * cHEADS);  float* ssP = carve((size_t)cNP * cHEADS);
    const int N = cNI + cNP;
    float* xh = carve((size_t)N * cHID);
    float* xg = carve(N); float* mg = carve(N); float* sg = carve(N); float* sc = carve(N);
    float* tv = carve(8); int* pidx = (int*)carve(8);
    float* hb = carve((size_t)cKPOOL * cHID);
    if ((size_t)((char*)W - (char*)d_ws) > ws_size) return; // insufficient scratch

    auto fill = [&](float* p, float v, size_t n) {
        k_fill<<<(unsigned)((n + 255) / 256), 256, 0, stream>>>(p, v, n);
    };
    auto copyk = [&](float* d, const float* s, size_t n) {
        k_copy<<<(unsigned)((n + 255) / 256), 256, 0, stream>>>(d, s, n);
    };
    auto gelu_ip = [&](const float* in, float* out, size_t n) {
        k_gelu<<<(unsigned)((n + 255) / 256), 256, 0, stream>>>(in, out, n);
    };
    auto gemm = [&](const float* X, int lda, const float* Wm, int ldw, const float* bias,
                    float* Y, int ldy, int M, int K, int Nout,
                    int nh, int xH, int wH, int yH) {
        dim3 g((unsigned)(((M + 15) / 16) * (Nout / 16)), (unsigned)nh);
        k_wmma_gemm<<<g, 32, 0, stream>>>(X, lda, Wm, ldw, bias, Y, ldy,
                                          M, K, Nout, 0, xH, wH, yH);
    };
    auto runEdge = [&](const int* src, const int* dst, int E, const float* ke, const float* ve,
                       const float* q, float* out, float* ms, float* ss, int Nd,
                       const float* prel) {
        fill(ms, 0.0f, (size_t)Nd * cHEADS);   // 0 bits == -inf in encoded order
        fill(ss, 0.0f, (size_t)Nd * cHEADS);
        unsigned eb = (unsigned)(((size_t)E * 32 + 255) / 256);
        k_edge_logits<<<eb, 256, 0, stream>>>(src, dst, E, ke, q, prel, cSCALE,
                                              al, (unsigned*)ms);
        k_edge_exp<<<(unsigned)(((size_t)E * 4 + 255) / 256), 256, 0, stream>>>(
            dst, E, al, (const unsigned*)ms, ss);
        k_edge_scatter<<<eb, 256, 0, stream>>>(src, dst, E, al, ss, ve, out);
    };

    // ---------------- 3 HGT layers ----------------
    for (int l = 0; l < 3; ++l) {
        const ConvP& P = conv[l];
        const float* XI = (l == 0) ? x_instr : xi; int ldi = (l == 0) ? 64 : 128;
        const float* XP = (l == 0) ? x_port  : xp_; int ldp = (l == 0) ? 32 : 128;
        // K/Q/V projections (WMMA)
        gemm(XI, ldi, P.k_w[0], cHID, P.k_b[0], ki, cHID, cNI, ldi, cHID, 1, 0, 0, 0);
        gemm(XI, ldi, P.q_w[0], cHID, P.q_b[0], qi, cHID, cNI, ldi, cHID, 1, 0, 0, 0);
        gemm(XI, ldi, P.v_w[0], cHID, P.v_b[0], vi, cHID, cNI, ldi, cHID, 1, 0, 0, 0);
        gemm(XP, ldp, P.k_w[1], cHID, P.k_b[1], kp, cHID, cNP, ldp, cHID, 1, 0, 0, 0);
        gemm(XP, ldp, P.q_w[1], cHID, P.q_b[1], qp, cHID, cNP, ldp, cHID, 1, 0, 0, 0);
        gemm(XP, ldp, P.v_w[1], cHID, P.v_b[1], vp, cHID, cNP, ldp, cHID, 1, 0, 0, 0);
        // per-NODE relation transforms, batched over heads (gridDim.y=4)
        gemm(ki, cHID, P.a_rel[0], 32, nullptr, keii, cHID, cNI, 32, 32, 4, 32, 1024, 32);
        gemm(vi, cHID, P.m_rel[0], 32, nullptr, veii, cHID, cNI, 32, 32, 4, 32, 1024, 32);
        gemm(ki, cHID, P.a_rel[1], 32, nullptr, keip, cHID, cNI, 32, 32, 4, 32, 1024, 32);
        gemm(vi, cHID, P.m_rel[1], 32, nullptr, veip, cHID, cNI, 32, 32, 4, 32, 1024, 32);
        gemm(kp, cHID, P.a_rel[2], 32, nullptr, kepi, cHID, cNP, 32, 32, 4, 32, 1024, 32);
        gemm(vp, cHID, P.m_rel[2], 32, nullptr, vepi, cHID, cNP, 32, 32, 4, 32, 1024, 32);
        // attention aggregation (segment softmax per relation, accumulate per dst type)
        fill(oi, 0.0f, SI); fill(op_, 0.0f, SP);
        runEdge(ei_ii, ei_ii + cEII, cEII, keii, veii, qi, oi, msI, ssI, cNI, P.p_rel[0]);
        runEdge(ei_ip, ei_ip + cEIP, cEIP, keip, veip, qp, op_, msP, ssP, cNP, P.p_rel[1]);
        runEdge(ei_pi, ei_pi + cEPI, cEPI, kepi, vepi, qi, oi, msI, ssI, cNI, P.p_rel[2]);
        // node update: gelu -> a_w projection -> gated skip (layers 1,2)
        gelu_ip(oi, oi, SI);
        gemm(oi, cHID, P.a_w[0], cHID, P.a_b[0], ti, cHID, cNI, cHID, cHID, 1, 0, 0, 0);
        gelu_ip(op_, op_, SP);
        gemm(op_, cHID, P.a_w[1], cHID, P.a_b[1], tp, cHID, cNP, cHID, cHID, 1, 0, 0, 0);
        if (l > 0) {
            k_skip<<<(unsigned)((SI + 255) / 256), 256, 0, stream>>>(ti, xi, P.skip[0], SI);
            k_skip<<<(unsigned)((SP + 255) / 256), 256, 0, stream>>>(tp, xp_, P.skip[1], SP);
        }
        // post: LN(+res at l==1)+gelu for l<2; plain gelu for l==2
        if (l < 2) {
            const float* rI = (l == 1) ? xi  : nullptr;  // pre-conv x (read-before-write)
            const float* rP = (l == 1) ? xp_ : nullptr;
            unsigned bi = (unsigned)(((size_t)cNI * 32 + 255) / 256);
            unsigned bp = (unsigned)(((size_t)cNP * 32 + 255) / 256);
            k_ln_gelu<<<bi, 256, 0, stream>>>(ti, nrm[0][l][1], nrm[0][l][0], rI, xi, cNI);
            k_ln_gelu<<<bp, 256, 0, stream>>>(tp, nrm[1][l][1], nrm[1][l][0], rP, xp_, cNP);
        } else {
            gelu_ip(ti, xi, SI);
            gelu_ip(tp, xp_, SP);
        }
    }

    // ---------------- SAGPool (GAT scoring) + head ----------------
    copyk(xh, xi, SI);
    copyk(xh + SI, xp_, SP);
    k_node_dot<<<(unsigned)(((size_t)N * 32 + 255) / 256), 256, 0, stream>>>(xh, g_w, xg, N);
    fill(mg, 0.0f, N); fill(sg, 0.0f, N); fill(sc, 0.0f, N);
    struct Chunk { const int* s; const int* d; int so, doo, E, loop; size_t off; };
    Chunk ch[4] = {
        { ei_ii, ei_ii + cEII, 0,   0,   cEII, 0, 0 },
        { ei_ip, ei_ip + cEIP, 0,   cNI, cEIP, 0, (size_t)cEII },
        { ei_pi, ei_pi + cEPI, cNI, 0,   cEPI, 0, (size_t)cEII + cEIP },
        { nullptr, nullptr,    0,   0,   N,    1, (size_t)cEII + cEIP + cEPI },
    };
    for (int i = 0; i < 4; ++i)
        k_gat_logit<<<(unsigned)((ch[i].E + 255) / 256), 256, 0, stream>>>(
            ch[i].s, ch[i].d, ch[i].so, ch[i].doo, ch[i].E, ch[i].loop,
            xg, g_asrc, g_adst, al + ch[i].off, (unsigned*)mg);
    for (int i = 0; i < 4; ++i)
        k_gat_exp<<<(unsigned)((ch[i].E + 255) / 256), 256, 0, stream>>>(
            ch[i].s, ch[i].d, ch[i].so, ch[i].doo, ch[i].E, ch[i].loop,
            al + ch[i].off, (const unsigned*)mg, sg);
    for (int i = 0; i < 4; ++i)
        k_gat_score<<<(unsigned)((ch[i].E + 255) / 256), 256, 0, stream>>>(
            ch[i].s, ch[i].d, ch[i].so, ch[i].doo, ch[i].E, ch[i].loop,
            xg, al + ch[i].off, sg, sc);
    k_topk8<<<1, 256, 0, stream>>>(sc, N, g_bias, pidx, tv);
    k_pool_gather<<<4, 256, 0, stream>>>(xh, pidx, tv, hb);
    k_head_mlp<<<1, 128, 0, stream>>>(hb, fc1_w, fc1_b, fc2_w, fc2_b, fc3_w, fc3_b,
                                      (float*)d_out);
}
